// unit_cgcn_1632087573205
// MI455X (gfx1250) — matrix operand order repaired and checked by reading, not compiled, not run on hardware
//
#include <hip/hip_runtime.h>
#include <hip/hip_bf16.h>

typedef __bf16 bf16;
typedef __attribute__((ext_vector_type(16))) __bf16 v16bf;
typedef __attribute__((ext_vector_type(8)))  float  v8f;
typedef __attribute__((ext_vector_type(4)))  unsigned int v4u;
typedef __attribute__((ext_vector_type(8)))  int v8i;
typedef __attribute__((ext_vector_type(4)))  int v4i;

#define NN     64
#define CIN    64
#define COUT   128
#define TT     256
#define VVV    25
#define PP     (TT * VVV)      // 6400 positions per (n, channel)
#define INTER  32
#define KLOG   (INTER * TT)    // 8192
#define KSPLIT 8               // K-parallelism for the logits Gram reduction
#define BNEPS  1e-5f

// ---------------------------------------------------------------------------
// fp32 -> bf16 conversion (grid-stride)
// ---------------------------------------------------------------------------
__global__ void cvt_f32_bf16(const float* __restrict__ src, bf16* __restrict__ dst, int n) {
  int stride = gridDim.x * blockDim.x;
  for (int i = blockIdx.x * blockDim.x + threadIdx.x; i < n; i += stride)
    dst[i] = (bf16)src[i];
}

// ---------------------------------------------------------------------------
// conv1x1 as GEMM:  out[n,o,p] = sum_c W[o,c] * X[n,c,p] + bias[o]
// X: (N, C, PP) bf16.  W: (O, C) bf16 row-major.  Block = 128 thr (4 waves),
// block tile = 16 O-rows x 64 P-cols; each wave owns one 16x16 WMMA tile.
// The strided 32x128B x-tile is staged into LDS by the Tensor Data Mover
// (one wave builds a 2D D# and issues tensor_load_to_lds) when the TDM
// builtin exists; otherwise falls back to cooperative vector loads.
// OUT: 0 = bf16 store, 1 = f32 store, 2 = f32 accumulate (+=).
// ---------------------------------------------------------------------------
template <int OUT>
__global__ __launch_bounds__(128) void conv1x1_wmma(const bf16* __restrict__ X,
                                                    const bf16* __restrict__ W,
                                                    const float* __restrict__ bias,
                                                    void* __restrict__ outp,
                                                    int C, int O) {
  __shared__ bf16 lds_w[16 * 64];   // 16 O-rows x C (C <= 64)
  __shared__ bf16 lds_x[32 * 64];   // 32 K-rows x 64 P-cols

  const int n = blockIdx.z, otile = blockIdx.y, ptile = blockIdx.x;
  const int tid = threadIdx.x, wave = tid >> 5, lane = tid & 31;
  const int m = lane & 15, hi = lane >> 4;

  // stage the 16xC weight tile once
  for (int i = tid; i < 16 * C; i += 128)
    lds_w[i] = W[(size_t)(otile * 16 + i / C) * C + (i % C)];

  const bf16* xn = X + (size_t)n * C * PP + (size_t)ptile * 64;

  v8f acc = {};
  for (int k0 = 0; k0 < C; k0 += 32) {
    __syncthreads();                      // lds_x free for reuse
#if __has_builtin(__builtin_amdgcn_tensor_load_to_lds) && \
    __has_builtin(__builtin_amdgcn_s_wait_tensorcnt)
    if (wave == 0) {
      // ---- Tensor DMA descriptor (D#), 2D tile: 32 rows x 64 bf16 ----
      unsigned lds_off = (unsigned)(size_t)(&lds_x[0]);     // LDS aperture: low 32 bits
      unsigned long long ga = (unsigned long long)(size_t)(xn + (size_t)k0 * PP);
      v4u g0;
      g0.x = 1u;                                            // count=1, user mode
      g0.y = lds_off;                                       // lds_addr [63:32]
      g0.z = (unsigned)(ga & 0xFFFFFFFFull);                // global_addr [95:64]
      g0.w = (unsigned)((ga >> 32) & 0x1FFFFFFull)          // global_addr [120:96]
             | (2u << 30);                                  // type=2 ("image") [127:126]
      v8i g1;
      g1[0] = (int)(1u << 16);                              // data_size=2B; no flags
      g1[1] = (int)((PP & 0xFFFF) << 16);                   // tensor_dim0[15:0] @ [31:16]
      g1[2] = (int)(((PP >> 16) & 0xFFFF) | (64 << 16));    // dim0 hi | tensor_dim1=64 lo
      g1[3] = (int)(64 << 16);                              // dim1 hi=0 | tile_dim0=64
      g1[4] = 32;                                           // tile_dim1=32, tile_dim2=0
      g1[5] = (int)PP;                                      // tensor_dim0_stride lo
      g1[6] = 0;                                            // stride hi | dim1_stride lo
      g1[7] = 0;
      v4i gz = {0, 0, 0, 0};                                // groups 2/3: dims 3/4 unused
      v8i gz8 = {0, 0, 0, 0, 0, 0, 0, 0};                   // extra group (6-arg form)
      __builtin_amdgcn_tensor_load_to_lds(g0, g1, gz, gz, gz8, 0);
      __builtin_amdgcn_s_wait_tensorcnt(0);
    }
#else
    if (k0 + 32 < C)  // hint next K-slab into cache (global_prefetch_b8)
      __builtin_prefetch(xn + (size_t)(k0 + 32) * PP, 0, 1);
    for (int i = tid; i < 32 * 64; i += 128) {
      int kk = i >> 6, cc = i & 63;                 // coalesced 64-col rows
      lds_x[i] = xn[(size_t)(k0 + kk) * PP + cc];
    }
#endif
    __syncthreads();

    v16bf a, b;
#pragma unroll
    for (int e = 0; e < 16; ++e) {
      int ka = e + 16 * hi;                         // A 16x32 bf16 layout
      a[e] = lds_w[m * C + k0 + ka];
      int kb = e + 16 * hi;                         // B 32x16 bf16 layout
      b[e] = lds_x[kb * 64 + wave * 16 + m];
    }
    acc = __builtin_amdgcn_wmma_f32_16x16x32_bf16(false, a, false, b,
                                                  (short)0, acc, false, false);
  }

  const int col = ptile * 64 + wave * 16 + m;
#pragma unroll
  for (int r = 0; r < 8; ++r) {
    int o = otile * 16 + r + 8 * hi;                // C/D layout: M = r + 8*hi
    float val = acc[r] + bias[o];
    size_t idx = (size_t)n * O * PP + (size_t)o * PP + col;
    if (OUT == 0)       ((bf16*)outp)[idx] = (bf16)val;
    else if (OUT == 1)  ((float*)outp)[idx] = val;
    else                ((float*)outp)[idx] += val;
  }
}

// ---------------------------------------------------------------------------
// logits[n,v,w] += (1/8192) * sum_{k in my K-slice} a[n,k,v] * b[n,k,w]
// a,b: (N, 8192, 25).  grid (N, KSPLIT); K blocked into 128-row LDS tiles;
// each thread owns <=3 of the 625 (v,w) pairs; partials combined with
// f32 atomics (L zeroed beforehand).
// ---------------------------------------------------------------------------
__global__ __launch_bounds__(256) void logits_kernel(const bf16* __restrict__ A,
                                                     const bf16* __restrict__ B,
                                                     float* __restrict__ L) {
  __shared__ bf16 la[128 * 26];
  __shared__ bf16 lb[128 * 26];
  const int n = blockIdx.x, tid = threadIdx.x;
  const int kbeg = blockIdx.y * (KLOG / KSPLIT);
  const int kend = kbeg + (KLOG / KSPLIT);
  const size_t base = (size_t)n * KLOG * VVV;

  int p0 = tid, p1 = tid + 256, p2 = tid + 512;
  int v0 = p0 / 25, w0 = p0 % 25;
  int v1 = p1 / 25, w1 = p1 % 25;
  bool has2 = (p2 < 625);
  int v2 = has2 ? p2 / 25 : 0, w2 = has2 ? p2 % 25 : 0;
  float acc0 = 0.f, acc1 = 0.f, acc2 = 0.f;

  for (int k0 = kbeg; k0 < kend; k0 += 128) {
    __syncthreads();
    for (int i = tid; i < 128 * 25; i += 256) {
      int r = i / 25, c = i % 25;
      la[r * 26 + c] = A[base + (size_t)(k0 + r) * 25 + c];
      lb[r * 26 + c] = B[base + (size_t)(k0 + r) * 25 + c];
    }
    __syncthreads();
    for (int kk = 0; kk < 128; ++kk) {
      acc0 += (float)la[kk * 26 + v0] * (float)lb[kk * 26 + w0];
      acc1 += (float)la[kk * 26 + v1] * (float)lb[kk * 26 + w1];
      acc2 += (float)la[kk * 26 + v2] * (float)lb[kk * 26 + w2];
    }
  }
  const float sc = 1.0f / (float)KLOG;
  atomicAdd(&L[n * 625 + p0], acc0 * sc);
  atomicAdd(&L[n * 625 + p1], acc1 * sc);
  if (has2) atomicAdd(&L[n * 625 + p2], acc2 * sc);
}

// ---------------------------------------------------------------------------
// S = softmax(logits, axis=v) + (A + PA)[subset], stored bf16 zero-padded to
// 32x32 per n (ready for WMMA K/N padding).  grid (N, 32), block 32.
// ---------------------------------------------------------------------------
__global__ void softmax_addA(const float* __restrict__ L, const float* __restrict__ Aadj,
                             const float* __restrict__ PA, int subset,
                             bf16* __restrict__ Sh) {
  const int n = blockIdx.x, w = blockIdx.y, v = threadIdx.x;
  const bool valid = (v < VVV) && (w < VVV);
  float x = valid ? L[n * 625 + v * 25 + w] : -1e30f;
  float mx = x;
  for (int off = 16; off > 0; off >>= 1) mx = fmaxf(mx, __shfl_xor(mx, off));
  float e = valid ? __expf(x - mx) : 0.f;
  float s = e;
  for (int off = 16; off > 0; off >>= 1) s += __shfl_xor(s, off);
  float out = 0.f;
  if (valid) {
    int ai = subset * 625 + v * 25 + w;
    out = e / s + Aadj[ai] + PA[ai];
  }
  Sh[(size_t)n * 1024 + v * 32 + w] = (bf16)out;
}

// ---------------------------------------------------------------------------
// agg[n,q,w] = sum_v x_flat[n,q,v] * S[n,v,w]  (q = c*T+t, 16384 rows per n)
// WMMA with K=25 zero-padded to 32, N=25 covered by two 16-col tiles.
// grid (256, N); 4 waves/block, one 16-row q tile per wave.
// ---------------------------------------------------------------------------
__global__ __launch_bounds__(128) void agg_wmma(const bf16* __restrict__ Xh,
                                                const bf16* __restrict__ Sh,
                                                bf16* __restrict__ Agg) {
  __shared__ bf16 s_s[1024];       // padded 32x32 S
  __shared__ bf16 s_a[4][512];     // per-wave 16x32 padded x tile
  const int n = blockIdx.y;
  const int tid = threadIdx.x, wave = tid >> 5, lane = tid & 31;
  const int m = lane & 15, hi = lane >> 4;

  for (int i = tid; i < 1024; i += 128) s_s[i] = Sh[(size_t)n * 1024 + i];
  __syncthreads();

  v16bf b0, b1;
#pragma unroll
  for (int e = 0; e < 16; ++e) {
    int k = e + 16 * hi;
    b0[e] = s_s[k * 32 + m];
    b1[e] = s_s[k * 32 + 16 + m];
  }

  const int qt = blockIdx.x * 4 + wave;                    // 0..1023
  const bf16* xrow = Xh + (size_t)n * CIN * PP + (size_t)qt * 16 * VVV;
  bf16* sa = s_a[wave];
  for (int i = lane; i < 512; i += 32) {                   // wave-private stage
    int r = i >> 5, c = i & 31;
    sa[i] = (c < VVV) ? xrow[r * VVV + c] : (bf16)0.f;
  }
  v16bf a;
#pragma unroll
  for (int e = 0; e < 16; ++e) {
    int ka = e + 16 * hi;
    a[e] = sa[m * 32 + ka];
  }
  v8f c0 = {}, c1 = {};
  c0 = __builtin_amdgcn_wmma_f32_16x16x32_bf16(false, a, false, b0, (short)0, c0, false, false);
  c1 = __builtin_amdgcn_wmma_f32_16x16x32_bf16(false, a, false, b1, (short)0, c1, false, false);

  bf16* outb = Agg + (size_t)n * CIN * PP;
#pragma unroll
  for (int r = 0; r < 8; ++r) {
    int q = qt * 16 + r + 8 * hi;
    outb[(size_t)q * VVV + m] = (bf16)c0[r];
    if (m < VVV - 16) outb[(size_t)q * VVV + 16 + m] = (bf16)c1[r];
  }
}

// ---------------------------------------------------------------------------
// Per-channel sum / sum-of-squares: one block per (n, channel) slice of PP.
// ---------------------------------------------------------------------------
__global__ __launch_bounds__(256) void chan_stats(const float* __restrict__ data,
                                                  float* __restrict__ sums,
                                                  float* __restrict__ sumsq) {
  const int b = blockIdx.x;               // n*COUT + o
  const int o = b & (COUT - 1);
  const size_t base = (size_t)b * PP;
  const int tid = threadIdx.x;
  float s = 0.f, q = 0.f;
  for (int i = tid; i < PP; i += 256) {
    float v = data[base + i];
    s += v; q += v * v;
  }
  for (int off = 16; off > 0; off >>= 1) { s += __shfl_xor(s, off); q += __shfl_xor(q, off); }
  __shared__ float ss[8], sq[8];
  if ((tid & 31) == 0) { ss[tid >> 5] = s; sq[tid >> 5] = q; }
  __syncthreads();
  if (tid == 0) {
    float S = 0.f, Q = 0.f;
    for (int wv = 0; wv < 8; ++wv) { S += ss[wv]; Q += sq[wv]; }
    atomicAdd(&sums[o], S);
    atomicAdd(&sumsq[o], Q);
  }
}

// ---------------------------------------------------------------------------
// out = BN(y) + BN(down);  down is already resident in `out`.
// stats layout: [sumY 128][sqY 128][sumD 128][sqD 128]
// ---------------------------------------------------------------------------
__global__ __launch_bounds__(256) void finalize_bn(const float* __restrict__ y,
                                                   float* __restrict__ out,
                                                   const float* __restrict__ st,
                                                   const float* __restrict__ gbn,
                                                   const float* __restrict__ bbn,
                                                   const float* __restrict__ gdn,
                                                   const float* __restrict__ bdn) {
  const float inv = 1.0f / (float)((size_t)NN * PP);
  const size_t total = (size_t)NN * COUT * PP;
  for (size_t i = blockIdx.x * (size_t)blockDim.x + threadIdx.x; i < total;
       i += (size_t)gridDim.x * blockDim.x) {
    int o = (int)((i / PP) & (COUT - 1));
    float my = st[o] * inv;       float vy = st[128 + o] * inv - my * my;
    float md = st[256 + o] * inv; float vd = st[384 + o] * inv - md * md;
    float yy = (y[i] - my) * rsqrtf(vy + BNEPS) * gbn[o] + bbn[o];
    float dd = (out[i] - md) * rsqrtf(vd + BNEPS) * gdn[o] + bdn[o];
    out[i] = yy + dd;
  }
}

// ---------------------------------------------------------------------------
extern "C" void kernel_launch(void* const* d_in, const int* in_sizes, int n_in,
                              void* d_out, int out_size, void* d_ws, size_t ws_size,
                              hipStream_t stream) {
  (void)in_sizes; (void)n_in; (void)out_size; (void)ws_size;
  const float* x    = (const float*)d_in[0];
  const float* Aadj = (const float*)d_in[1];
  const float* PA   = (const float*)d_in[2];
  const float* Wa   = (const float*)d_in[3];
  const float* ba   = (const float*)d_in[4];
  const float* Wb   = (const float*)d_in[5];
  const float* bb   = (const float*)d_in[6];
  const float* Wd   = (const float*)d_in[7];
  const float* bd   = (const float*)d_in[8];
  const float* Wdw  = (const float*)d_in[9];
  const float* bdw  = (const float*)d_in[10];
  const float* gbn  = (const float*)d_in[11];
  const float* bbn  = (const float*)d_in[12];
  const float* gdn  = (const float*)d_in[13];
  const float* betd = (const float*)d_in[14];
  float* out = (float*)d_out;

  char* ws = (char*)d_ws;
  size_t off = 0;
  auto alloc = [&](size_t bytes) {
    size_t o = off; off += (bytes + 255) & ~(size_t)255; return o;
  };
  bf16*  xh    = (bf16*)(ws + alloc((size_t)NN * CIN * PP * 2));      // 52.4 MB
  bf16*  wah   = (bf16*)(ws + alloc((size_t)3 * INTER * CIN * 2));
  bf16*  wbh   = (bf16*)(ws + alloc((size_t)3 * INTER * CIN * 2));
  bf16*  wdh   = (bf16*)(ws + alloc((size_t)3 * COUT * CIN * 2));
  bf16*  wdwh  = (bf16*)(ws + alloc((size_t)COUT * CIN * 2));
  bf16*  abuf  = (bf16*)(ws + alloc((size_t)NN * INTER * PP * 2));    // 26.2 MB
  bf16*  bbuf  = (bf16*)(ws + alloc((size_t)NN * INTER * PP * 2));    // 26.2 MB
  bf16*  aggh  = (bf16*)(ws + alloc((size_t)NN * CIN * PP * 2));      // 52.4 MB
  float* ybuf  = (float*)(ws + alloc((size_t)NN * COUT * PP * 4));    // 209.7 MB
  float* lbuf  = (float*)(ws + alloc((size_t)NN * 625 * 4));
  bf16*  Shb   = (bf16*)(ws + alloc((size_t)NN * 1024 * 2));
  float* stats = (float*)(ws + alloc(512 * 4));

  // precision down-convert (memory-bound problem -> bf16 operands everywhere)
  cvt_f32_bf16<<<2048, 256, 0, stream>>>(x, xh, NN * CIN * PP);
  cvt_f32_bf16<<<16, 256, 0, stream>>>(Wa, wah, 3 * INTER * CIN);
  cvt_f32_bf16<<<16, 256, 0, stream>>>(Wb, wbh, 3 * INTER * CIN);
  cvt_f32_bf16<<<48, 256, 0, stream>>>(Wd, wdh, 3 * COUT * CIN);
  cvt_f32_bf16<<<16, 256, 0, stream>>>(Wdw, wdwh, COUT * CIN);

  const dim3 convGridI(PP / 64, INTER / 16, NN);   // conv_a / conv_b
  const dim3 convGridO(PP / 64, COUT / 16, NN);    // conv_d / conv_down

  for (int i = 0; i < 3; ++i) {
    conv1x1_wmma<0><<<convGridI, 128, 0, stream>>>(xh, wah + i * INTER * CIN,
                                                   ba + i * INTER, abuf, CIN, INTER);
    conv1x1_wmma<0><<<convGridI, 128, 0, stream>>>(xh, wbh + i * INTER * CIN,
                                                   bb + i * INTER, bbuf, CIN, INTER);
    (void)hipMemsetAsync(lbuf, 0, (size_t)NN * 625 * 4, stream);
    logits_kernel<<<dim3(NN, KSPLIT), 256, 0, stream>>>(abuf, bbuf, lbuf);
    softmax_addA<<<dim3(NN, 32), 32, 0, stream>>>(lbuf, Aadj, PA, i, Shb);
    agg_wmma<<<dim3(256, NN), 128, 0, stream>>>(xh, Shb, aggh);
    if (i == 0)
      conv1x1_wmma<1><<<convGridO, 128, 0, stream>>>(aggh, wdh + i * COUT * CIN,
                                                     bd + i * COUT, ybuf, CIN, COUT);
    else
      conv1x1_wmma<2><<<convGridO, 128, 0, stream>>>(aggh, wdh + i * COUT * CIN,
                                                     bd + i * COUT, ybuf, CIN, COUT);
  }

  // down branch straight into d_out (read back by finalize)
  conv1x1_wmma<1><<<convGridO, 128, 0, stream>>>(xh, wdwh, bdw, out, CIN, COUT);

  (void)hipMemsetAsync(stats, 0, 512 * 4, stream);
  chan_stats<<<NN * COUT, 256, 0, stream>>>(ybuf, stats, stats + 128);
  chan_stats<<<NN * COUT, 256, 0, stream>>>(out, stats + 256, stats + 384);
  finalize_bn<<<4096, 256, 0, stream>>>(ybuf, out, stats, gbn, bbn, gdn, betd);
}
